// MARN_10479720202424
// MI455X (gfx1250) — matrix operand order repaired and checked by reading, not compiled, not run on hardware
//
#include <hip/hip_runtime.h>
#include <hip/hip_bf16.h>

typedef _Float16 half_t;
typedef __attribute__((ext_vector_type(16))) _Float16 v16h;
typedef __attribute__((ext_vector_type(8)))  _Float16 v8h;
typedef __attribute__((ext_vector_type(8)))  float    v8f;

// ---------------- problem dims ----------------
constexpr int kDL = 300, kDA = 74, kDV = 35, kDX = 409;  // x split
constexpr int kXLP = 320, kXAP = 96, kXVP = 64;          // K-padded x widths
constexpr int kDHL = 128, kDHA = 32, kDHV = 32;
constexpr int kH   = 192;                                 // DH_L+DH_A+DH_V
constexpr int kRED = 64, kMAPH = 256, kHOUT = 64;
constexpr int kT = 128, kN = 2048;
constexpr int kRowsPerBlk = 16;

// ---------------- packed-weight offsets in workspace (half elements) -----
// pack layout per matrix: [tile][chunk][lane][16] ; n = tile*16+(lane&15),
// k = chunk*32 + (lane>>4)*16 + e  (zero-padded past real in_dim)
constexpr size_t OFF_WL  = 0;               // 512 x 320 -> 163840
constexpr size_t OFF_UL  = 163840;          // 512 x 128 ->  65536
constexpr size_t OFF_VL  = 229376;          // 512 x 192 ->  98304
constexpr size_t OFF_WA  = 327680;          // 128 x  96 ->  12288
constexpr size_t OFF_UA  = 339968;          // 128 x  32 ->   4096
constexpr size_t OFF_VA  = 344064;          // 128 x 192 ->  24576
constexpr size_t OFF_WV  = 368640;          // 128 x  64 ->   8192
constexpr size_t OFF_UV  = 376832;          // 128 x  32 ->   4096
constexpr size_t OFF_VV  = 380928;          // 128 x 192 ->  24576
constexpr size_t OFF_ATT = 405504;          // 768 x 192 -> 147456
constexpr size_t OFF_RL  = 552960;          //  32 x 512 ->  16384
constexpr size_t OFF_RA  = 569344;          //  16 x 128 ->   2048
constexpr size_t OFF_RV  = 571392;          //  16 x 128 ->   2048
constexpr size_t OFF_FC1 = 573440;          // 256 x  64 ->  16384
constexpr size_t OFF_FC2 = 589824;          // 192 x 256 ->  49152
constexpr size_t OFF_O1  = 638976;          //  64 x 384 ->  24576
constexpr size_t PACK_TOTAL = 663552;       // halves (= 1,327,104 bytes)

// ---------------- LDS carve (bytes) ----------------
constexpr int L_BIG = 0;         // f32 [16][768]  gates / attention scratch
constexpr int L_C   = 49152;     // f32 [16][192]  cell state (c_l|c_a|c_v)
constexpr int L_XL  = 61440;     // f16 [16][320]
constexpr int L_XA  = 71680;     // f16 [16][96]
constexpr int L_XV  = 74752;     // f16 [16][64]
constexpr int L_HL  = 76800;     // f16 [16][128]
constexpr int L_HA  = 80896;     // f16 [16][32]
constexpr int L_HV  = 81920;     // f16 [16][32]
constexpr int L_Z   = 82944;     // f16 [16][192]
constexpr int L_CS  = 89088;     // f16 [16][192]  new_cs
constexpr int L_ATD = 95232;     // f16 [16][768]  attended
constexpr int L_R   = 119808;    // f16 [16][64]
constexpr int L_M   = 121856;    // f16 [16][256]
constexpr int L_O   = 130048;    // f16 [16][64]
constexpr int LDS_BYTES = 132096;

struct MarnArgs {
  const float* x;
  const half_t* pk;
  const float *bWl, *bUl, *bVl;
  const float *bWa, *bUa, *bVa;
  const float *bWv, *bUv, *bVv;
  const float *att_b, *rl_b, *ra_b, *rv_b;
  const float *fc1_b, *fc2_b, *o1_b, *o2_w, *o2_b;
  float* out;
};

// ---------------- prep: pack a weight matrix (out x in, row-major) into
// B-fragment layout with K zero-padding -------------------------------
__global__ void pack_w(const float* __restrict__ src, half_t* __restrict__ dst,
                       int out_dim, int in_dim, int chunks) {
  int idx = blockIdx.x * 256 + threadIdx.x;
  int total = (out_dim / 16) * chunks * 512;
  if (idx >= total) return;
  int e     = idx & 15;
  int lane  = (idx >> 4) & 31;
  int chunk = (idx >> 9) % chunks;
  int tile  = (idx >> 9) / chunks;
  int n = tile * 16 + (lane & 15);
  int k = chunk * 32 + (lane >> 4) * 16 + e;
  float v = (k < in_dim) ? src[(size_t)n * in_dim + k] : 0.0f;
  dst[idx] = (half_t)v;
}

// ---------------- WMMA fragment helpers ----------------
// A 16x32 f16: lane<16 -> M=lane, K 0..15 ;
//              lane>=16 -> M=lane-16, K 16..31 (matches B pack order)
__device__ __forceinline__ v16h lds_a_frag(const half_t* base, int stride,
                                           int colbase, int lane) {
  int m  = lane & 15;
  int o0 = colbase + ((lane >> 4) << 4);
  const half_t* p = base + m * stride + o0;
  v8h lo = *(const v8h*)p;
  v8h hi = *(const v8h*)(p + 8);
  v16h r;
#pragma unroll
  for (int i = 0; i < 8; ++i) { r[i] = lo[i]; r[i + 8] = hi[i]; }
  return r;
}

__device__ __forceinline__ v16h gbl_b_frag(const half_t* __restrict__ pk,
                                           int chunks, int tile, int chunk,
                                           int lane) {
  const half_t* p = pk + ((size_t)(tile * chunks + chunk) * 32 + lane) * 16;
  return *(const v16h*)p;  // 32B contiguous per lane
}

// D 16x16 f32: lane<16 -> N=lane, M = vgpr ; lane>=16 -> M = 8+vgpr
__device__ __forceinline__ void store_tile_f(float* s, int stride, int colbase,
                                             int lane, v8f acc, float bias) {
  int n  = colbase + (lane & 15);
  int mb = (lane >> 4) * 8;
#pragma unroll
  for (int i = 0; i < 8; ++i) s[(mb + i) * stride + n] = acc[i] + bias;
}

__device__ __forceinline__ void store_tile_h(half_t* s, int stride, int colbase,
                                             int lane, v8f acc,
                                             const float* bias, int bbase,
                                             bool relu) {
  int n  = lane & 15;
  int mb = (lane >> 4) * 8;
  float b = bias[bbase + n];
#pragma unroll
  for (int i = 0; i < 8; ++i) {
    float v = acc[i] + b;
    if (relu) v = v > 0.0f ? v : 0.0f;
    s[(mb + i) * stride + colbase + n] = (half_t)v;
  }
}

// Batch-load all NT B fragments of a chunk into distinct registers before
// issuing WMMAs -> NT independent 32B loads in flight per wait, instead of
// one load / one wait / one wmma serial chain.
template <int CHUNKS, int NT>
__device__ __forceinline__ void gemm_seg(v8f* acc, const int* tiles,
                                         const half_t* abase, int astride,
                                         const half_t* __restrict__ pk,
                                         int lane) {
#pragma unroll
  for (int c = 0; c < CHUNKS; ++c) {
    v16h b[NT];
#pragma unroll
    for (int g = 0; g < NT; ++g)
      b[g] = gbl_b_frag(pk, CHUNKS, tiles[g], c, lane);
    v16h a = lds_a_frag(abase, astride, c * 32, lane);
#pragma unroll
    for (int g = 0; g < NT; ++g)
      acc[g] = __builtin_amdgcn_wmma_f32_16x16x32_f16(
          false, a, false, b[g], (short)0, acc[g], false, false);
  }
}

__device__ __forceinline__ float sigmoidf_(float x) {
  return 1.0f / (1.0f + __expf(-x));
}

// ---------------- main persistent-state kernel: one block = 16 batch rows,
// scans all 128 time steps ----------------
__launch_bounds__(256)
__global__ void marn_main(MarnArgs A) {
  extern __shared__ __align__(16) char smem[];
  float*  sBig = (float*)(smem + L_BIG);
  float*  sC   = (float*)(smem + L_C);
  half_t* sXL  = (half_t*)(smem + L_XL);
  half_t* sXA  = (half_t*)(smem + L_XA);
  half_t* sXV  = (half_t*)(smem + L_XV);
  half_t* sHL  = (half_t*)(smem + L_HL);
  half_t* sHA  = (half_t*)(smem + L_HA);
  half_t* sHV  = (half_t*)(smem + L_HV);
  half_t* sZ   = (half_t*)(smem + L_Z);
  half_t* sCS  = (half_t*)(smem + L_CS);
  half_t* sATD = (half_t*)(smem + L_ATD);
  half_t* sR   = (half_t*)(smem + L_R);
  half_t* sM   = (half_t*)(smem + L_M);
  half_t* sO   = (half_t*)(smem + L_O);

  const int tid  = threadIdx.x;
  const int wave = tid >> 5;
  const int lane = tid & 31;
  const int rowBase = blockIdx.x * kRowsPerBlk;

  // ---- init recurrent state + zero-pad x buffers ----
  for (int i = tid; i < 16 * kH;   i += 256) sC[i] = 0.0f;
  for (int i = tid; i < 16 * kDHL; i += 256) sHL[i] = (half_t)0.0f;
  for (int i = tid; i < 16 * 32;   i += 256) { sHA[i] = (half_t)0.0f; sHV[i] = (half_t)0.0f; }
  for (int i = tid; i < 16 * kH;   i += 256) sZ[i]  = (half_t)0.0f;
  for (int i = tid; i < 16 * kXLP; i += 256) sXL[i] = (half_t)0.0f;
  for (int i = tid; i < 16 * kXAP; i += 256) sXA[i] = (half_t)0.0f;
  for (int i = tid; i < 16 * kXVP; i += 256) sXV[i] = (half_t)0.0f;
  __syncthreads();

  for (int t = 0; t < kT; ++t) {
    // ---- stage x[t] tile into LDS (f32 -> f16, split + K-pad) ----
    const float* xt = A.x + ((size_t)t * kN + rowBase) * kDX;
    if (t + 1 < kT) {
      const float* xn = A.x + ((size_t)(t + 1) * kN + rowBase) * kDX;
      for (int i = tid * 16; i < 16 * kDX; i += 256 * 16)
        __builtin_prefetch(xn + i, 0, 1);
    }
    for (int i = tid; i < 16 * kDX; i += 256) {
      int r = i / kDX, c = i - r * kDX;
      half_t v = (half_t)xt[(size_t)r * kDX + c];
      if (c < kDL)            sXL[r * kXLP + c] = v;
      else if (c < kDL + kDA) sXA[r * kXAP + (c - kDL)] = v;
      else                    sXV[r * kXVP + (c - kDL - kDA)] = v;
    }
    __syncthreads();

    // ---- LSTHM-L gates: S = Xl@Wl^T + Hl@Ul^T + Z@Vl^T  (16x512) ----
    {
      v8f acc[4] = {};
      int tiles[4];
#pragma unroll
      for (int g = 0; g < 4; ++g) tiles[g] = g * 8 + wave;
      gemm_seg<10, 4>(acc, tiles, sXL, kXLP, A.pk + OFF_WL, lane);
      gemm_seg<4, 4>(acc, tiles, sHL, kDHL, A.pk + OFF_UL, lane);
      gemm_seg<6, 4>(acc, tiles, sZ, kH, A.pk + OFF_VL, lane);
#pragma unroll
      for (int g = 0; g < 4; ++g)
        store_tile_f(sBig, 768, tiles[g] * 16, lane, acc[g], 0.0f);
    }
    __syncthreads();

    // ---- pointwise L: gates -> c_l, h_l ----
    for (int i = tid; i < 16 * kDHL; i += 256) {
      int r = i >> 7, c = i & 127;
      const float* sr = sBig + r * 768;
      float f  = sigmoidf_(sr[c]       + A.bWl[c]       + A.bUl[c]       + A.bVl[c]);
      float ii = sigmoidf_(sr[128 + c] + A.bWl[128 + c] + A.bUl[128 + c] + A.bVl[128 + c]);
      float o  = sigmoidf_(sr[256 + c] + A.bWl[256 + c] + A.bUl[256 + c] + A.bVl[256 + c]);
      float g  = tanhf(sr[384 + c] + A.bWl[384 + c] + A.bUl[384 + c] + A.bVl[384 + c]);
      float cn = f * sC[r * kH + c] + ii * g;
      sC[r * kH + c]   = cn;
      sHL[r * kDHL + c] = (half_t)(tanhf(cn) * o);
      sCS[r * kH + c]   = (half_t)cn;
    }
    __syncthreads();

    // ---- LSTHM-A and LSTHM-V gates (16x128 each) ----
    {
      v8f accA[1] = {}, accV[1] = {};
      int tw[1] = { wave };
      gemm_seg<3, 1>(accA, tw, sXA, kXAP, A.pk + OFF_WA, lane);
      gemm_seg<1, 1>(accA, tw, sHA, 32, A.pk + OFF_UA, lane);
      gemm_seg<6, 1>(accA, tw, sZ, kH, A.pk + OFF_VA, lane);
      gemm_seg<2, 1>(accV, tw, sXV, kXVP, A.pk + OFF_WV, lane);
      gemm_seg<1, 1>(accV, tw, sHV, 32, A.pk + OFF_UV, lane);
      gemm_seg<6, 1>(accV, tw, sZ, kH, A.pk + OFF_VV, lane);
      store_tile_f(sBig, 768, wave * 16, lane, accA[0], 0.0f);
      store_tile_f(sBig, 768, 128 + wave * 16, lane, accV[0], 0.0f);
    }
    __syncthreads();

    // ---- pointwise A & V ----
    for (int i = tid; i < 16 * 64; i += 256) {
      int r = i >> 6, c2 = i & 63;
      const float* sr = sBig + r * 768;
      if (c2 < 32) {
        int c = c2;
        float f  = sigmoidf_(sr[c]      + A.bWa[c]      + A.bUa[c]      + A.bVa[c]);
        float ii = sigmoidf_(sr[32 + c] + A.bWa[32 + c] + A.bUa[32 + c] + A.bVa[32 + c]);
        float o  = sigmoidf_(sr[64 + c] + A.bWa[64 + c] + A.bUa[64 + c] + A.bVa[64 + c]);
        float g  = tanhf(sr[96 + c] + A.bWa[96 + c] + A.bUa[96 + c] + A.bVa[96 + c]);
        float cn = f * sC[r * kH + 128 + c] + ii * g;
        sC[r * kH + 128 + c] = cn;
        sHA[r * 32 + c]      = (half_t)(tanhf(cn) * o);
        sCS[r * kH + 128 + c] = (half_t)cn;
      } else {
        int c = c2 - 32;
        float f  = sigmoidf_(sr[128 + c] + A.bWv[c]      + A.bUv[c]      + A.bVv[c]);
        float ii = sigmoidf_(sr[160 + c] + A.bWv[32 + c] + A.bUv[32 + c] + A.bVv[32 + c]);
        float o  = sigmoidf_(sr[192 + c] + A.bWv[64 + c] + A.bUv[64 + c] + A.bVv[64 + c]);
        float g  = tanhf(sr[224 + c] + A.bWv[96 + c] + A.bUv[96 + c] + A.bVv[96 + c]);
        float cn = f * sC[r * kH + 160 + c] + ii * g;
        sC[r * kH + 160 + c] = cn;
        sHV[r * 32 + c]      = (half_t)(tanhf(cn) * o);
        sCS[r * kH + 160 + c] = (half_t)cn;
      }
    }
    __syncthreads();

    // ---- attention: att = new_cs @ att_w^T + att_b  (16x768) ----
    {
      v8f acc[6] = {};
      int tiles[6];
#pragma unroll
      for (int j = 0; j < 6; ++j) tiles[j] = wave + 8 * j;
      gemm_seg<6, 6>(acc, tiles, sCS, kH, A.pk + OFF_ATT, lane);
#pragma unroll
      for (int j = 0; j < 6; ++j) {
        int col = tiles[j] * 16 + (lane & 15);
        store_tile_f(sBig, 768, tiles[j] * 16, lane, acc[j], A.att_b[col]);
      }
    }
    __syncthreads();

    // ---- softmax over H per (row, k) group + attended = attn * cs ----
    if (tid < 64) {
      int r = tid >> 2, k = tid & 3;
      const float* ar = sBig + r * 768 + k * kH;
      float mx = ar[0];
      for (int h = 1; h < kH; ++h) mx = fmaxf(mx, ar[h]);
      float s = 0.0f;
      for (int h = 0; h < kH; ++h) s += __expf(ar[h] - mx);
      float inv = 1.0f / s;
      half_t* dr = sATD + r * 768 + k * kH;
      const float* cr = sC + r * kH;
      for (int h = 0; h < kH; ++h)
        dr[h] = (half_t)(__expf(ar[h] - mx) * inv * cr[h]);
    }
    __syncthreads();

    // ---- reduce GEMMs: r = [red_l | red_a | red_v]  (16x64) ----
    if (wave < 2) {
      v8f acc = {};
#pragma unroll
      for (int c = 0; c < 16; ++c) {
        int colb = (c >> 2) * kH + (c & 3) * 32;  // group k, 128-wide L slice
        v16h b = gbl_b_frag(A.pk + OFF_RL, 16, wave, c, lane);
        v16h a = lds_a_frag(sATD, 768, colb, lane);
        acc = __builtin_amdgcn_wmma_f32_16x16x32_f16(false, a, false, b,
                                                     (short)0, acc, false, false);
      }
      store_tile_h(sR, kRED, wave * 16, lane, acc, A.rl_b, wave * 16, false);
    } else if (wave == 2) {
      v8f acc = {};
#pragma unroll
      for (int c = 0; c < 4; ++c) {
        v16h b = gbl_b_frag(A.pk + OFF_RA, 4, 0, c, lane);
        v16h a = lds_a_frag(sATD, 768, c * kH + 128, lane);
        acc = __builtin_amdgcn_wmma_f32_16x16x32_f16(false, a, false, b,
                                                     (short)0, acc, false, false);
      }
      store_tile_h(sR, kRED, 32, lane, acc, A.ra_b, 0, false);
    } else if (wave == 3) {
      v8f acc = {};
#pragma unroll
      for (int c = 0; c < 4; ++c) {
        v16h b = gbl_b_frag(A.pk + OFF_RV, 4, 0, c, lane);
        v16h a = lds_a_frag(sATD, 768, c * kH + 160, lane);
        acc = __builtin_amdgcn_wmma_f32_16x16x32_f16(false, a, false, b,
                                                     (short)0, acc, false, false);
      }
      store_tile_h(sR, kRED, 48, lane, acc, A.rv_b, 0, false);
    }
    __syncthreads();

    // ---- fc1: relu(r @ fc1^T)  (16x256) ----
    {
      v8f acc[2] = {};
      int tiles[2] = { wave, wave + 8 };
      gemm_seg<2, 2>(acc, tiles, sR, kRED, A.pk + OFF_FC1, lane);
      store_tile_h(sM, kMAPH, wave * 16, lane, acc[0], A.fc1_b, wave * 16, true);
      store_tile_h(sM, kMAPH, (wave + 8) * 16, lane, acc[1], A.fc1_b,
                   (wave + 8) * 16, true);
    }
    __syncthreads();

    // ---- fc2: z = m @ fc2^T + b  (16x192) ----
    {
      v8f a0 = {}, a1 = {};
#pragma unroll
      for (int c = 0; c < 8; ++c) {
        v16h b0 = gbl_b_frag(A.pk + OFF_FC2, 8, wave, c, lane);
        v16h b1 = (wave < 4)
                      ? gbl_b_frag(A.pk + OFF_FC2, 8, wave + 8, c, lane)
                      : b0;
        v16h a = lds_a_frag(sM, kMAPH, c * 32, lane);
        a0 = __builtin_amdgcn_wmma_f32_16x16x32_f16(false, a, false, b0,
                                                    (short)0, a0, false, false);
        if (wave < 4)
          a1 = __builtin_amdgcn_wmma_f32_16x16x32_f16(false, a, false, b1,
                                                      (short)0, a1, false, false);
      }
      store_tile_h(sZ, kH, wave * 16, lane, a0, A.fc2_b, wave * 16, false);
      if (wave < 4)
        store_tile_h(sZ, kH, (wave + 8) * 16, lane, a1, A.fc2_b,
                     (wave + 8) * 16, false);
    }
    __syncthreads();
  }

  // ---- output head: relu([h_l|h_a|h_v|z] @ o1^T) @ o2^T ----
  if (wave < 4) {
    v8f acc = {};
#pragma unroll
    for (int c = 0; c < 12; ++c) {
      const half_t* ab;
      int as, colb;
      if (c < 4)       { ab = sHL; as = kDHL; colb = c * 32; }
      else if (c == 4) { ab = sHA; as = 32;   colb = 0; }
      else if (c == 5) { ab = sHV; as = 32;   colb = 0; }
      else             { ab = sZ;  as = kH;   colb = (c - 6) * 32; }
      v16h b = gbl_b_frag(A.pk + OFF_O1, 12, wave, c, lane);
      v16h a = lds_a_frag(ab, as, colb, lane);
      acc = __builtin_amdgcn_wmma_f32_16x16x32_f16(false, a, false, b,
                                                   (short)0, acc, false, false);
    }
    store_tile_h(sO, kHOUT, wave * 16, lane, acc, A.o1_b, wave * 16, true);
  }
  __syncthreads();
  if (tid < 16) {
    float s = A.o2_b[0];
    for (int k2 = 0; k2 < kHOUT; ++k2)
      s += (float)sO[tid * kHOUT + k2] * A.o2_w[k2];
    A.out[rowBase + tid] = s;
  }
}

// ---------------- launch ----------------
extern "C" void kernel_launch(void* const* d_in, const int* in_sizes, int n_in,
                              void* d_out, int out_size, void* d_ws, size_t ws_size,
                              hipStream_t stream) {
  (void)in_sizes; (void)n_in; (void)out_size; (void)ws_size;
  half_t* pk = (half_t*)d_ws;

  auto pack = [&](int idx, size_t off, int outd, int ind, int chunks) {
    int total = (outd / 16) * chunks * 512;
    pack_w<<<(total + 255) / 256, 256, 0, stream>>>(
        (const float*)d_in[idx], pk + off, outd, ind, chunks);
  };
  pack(1,  OFF_WL,  512, 300, 10);
  pack(3,  OFF_UL,  512, 128, 4);
  pack(5,  OFF_VL,  512, 192, 6);
  pack(7,  OFF_WA,  128, 74,  3);
  pack(9,  OFF_UA,  128, 32,  1);
  pack(11, OFF_VA,  128, 192, 6);
  pack(13, OFF_WV,  128, 35,  2);
  pack(15, OFF_UV,  128, 32,  1);
  pack(17, OFF_VV,  128, 192, 6);
  pack(19, OFF_ATT, 768, 192, 6);
  pack(21, OFF_RL,  32,  512, 16);
  pack(23, OFF_RA,  16,  128, 4);
  pack(25, OFF_RV,  16,  128, 4);
  pack(27, OFF_FC1, 256, 64,  2);
  pack(29, OFF_FC2, 192, 256, 8);
  pack(31, OFF_O1,  64,  384, 12);

  MarnArgs A;
  A.x    = (const float*)d_in[0];
  A.pk   = pk;
  A.bWl  = (const float*)d_in[2];  A.bUl = (const float*)d_in[4];  A.bVl = (const float*)d_in[6];
  A.bWa  = (const float*)d_in[8];  A.bUa = (const float*)d_in[10]; A.bVa = (const float*)d_in[12];
  A.bWv  = (const float*)d_in[14]; A.bUv = (const float*)d_in[16]; A.bVv = (const float*)d_in[18];
  A.att_b = (const float*)d_in[20];
  A.rl_b  = (const float*)d_in[22];
  A.ra_b  = (const float*)d_in[24];
  A.rv_b  = (const float*)d_in[26];
  A.fc1_b = (const float*)d_in[28];
  A.fc2_b = (const float*)d_in[30];
  A.o1_b  = (const float*)d_in[32];
  A.o2_w  = (const float*)d_in[33];
  A.o2_b  = (const float*)d_in[34];
  A.out   = (float*)d_out;

  marn_main<<<kN / kRowsPerBlk, 256, LDS_BYTES, stream>>>(A);
}
